// Shift_gcn_57097295233324
// MI455X (gfx1250) — matrix operand order repaired and checked by reading, not compiled, not run on hardware
//
#include <hip/hip_runtime.h>
#include <hip/hip_bf16.h>
#include <cstdint>

// ---------------------------------------------------------------------------
// Shift-GCN block for MI455X (gfx1250, wave32, WMMA).
//   pass A: z = gemm_bf16(gathered/scaled x0, W)  -> per-(v,d) partial sums
//   finalize: mean / rstd per feature
//   pass B: recompute gemm, apply BN (+shift_out perm), residual, relu, store
// Roofline: 49.4 GFLOP bf16-WMMA vs ~1.2 GB total HBM traffic @23.3TB/s ->
// memory bound; recomputing the GEMM beats storing the 386MB intermediate.
// This revision: saddr-form batched gathers, double-buffered A/B fragments,
// async global->LDS staging of W (ASYNCcnt path), fully unrolled WMMA chain.
// ---------------------------------------------------------------------------

typedef __bf16 v16bf __attribute__((ext_vector_type(16)));
typedef float  v8f   __attribute__((ext_vector_type(8)));

#define NSAMP 16384
#define NV    23
#define NC    256
#define NF    (NV * NC)      // 5888 features
#define MBLK  128            // rows (samples) per block
#define NMB   (NSAMP / MBLK) // 128 m-blocks per v

// workspace byte offsets
#define WS_OFF    0          // int[5888]  gather byte-offsets into one x0 slice
#define WS_SCL    23552      // f32[5888]  tanh(mask)+1
#define WS_INV    47104      // int[5888]  inverse shift_out
#define WS_MEAN   70656      // f32[5888]
#define WS_RSTD   94208      // f32[5888]
#define WS_WSW    117760     // bf16[65536] W pre-swizzled into B-fragments
#define WS_PSUM   248832     // f32[23*128*256] per-block column sums
#define WS_PSQ    3263488    // f32[23*128*256] per-block column sumsq

// LDS layout (dynamic)
#define L_W     0            // 131072 B : full B-matrix fragments
#define L_KOFF  131072       // 1024 B   : int[256] gather offsets (this v)
#define L_KSCL  132096       // 1024 B   : f32[256] scales (this v)
#define L_EPI   133120       // pass A: f32[8*256] x2 ; pass B: 2xf32[256]+int[256]
#define SMEM_A  (133120 + 8192 + 8192)        // 149504
#define SMEM_B  (133120 + 1024 + 1024 + 1024) // 136192

__global__ __launch_bounds__(256) void k_prep_tables(
    const float* __restrict__ mask, const int* __restrict__ shift_in,
    const int* __restrict__ shift_out, int* __restrict__ off_tab,
    float* __restrict__ scl_tab, int* __restrict__ inv_so)
{
    int f = blockIdx.x * 256 + threadIdx.x;   // 0..5887 exact
    int p = shift_in[f];
    // x_flat[n][w*C + c] = x0[n][c][w]  ->  byte offset inside one n-slice
    off_tab[f] = ((p & 255) * NV + (p >> 8)) * 4;
    scl_tab[f] = tanhf(mask[f]) + 1.0f;
    inv_so[shift_out[f]] = f;                 // permutation -> race-free
}

// Pre-swizzle W (256x256 f32, row-major [k][d]) into per-lane contiguous
// bf16 B-fragments: frag(kt 0..7, nt 0..15), lane holds col=nt*16+(lane&15),
// K = (lane>>4)*8 + [0..7], then 16 + (lane>>4)*8 + [0..7], dword v = pair (2v, 2v+1).
__global__ __launch_bounds__(256) void k_prep_w(
    const float* __restrict__ W, __bf16* __restrict__ wsw)
{
    int t    = blockIdx.x * 256 + threadIdx.x;  // 0..32767 exact (one dword each)
    int v    = t & 7;
    int lane = (t >> 3) & 31;
    int nt   = (t >> 8) & 15;
    int kt   = t >> 12;
    int col  = nt * 16 + (lane & 15);
    int kb   = kt * 32 + ((lane >> 4) << 3) + 2 * v + (v >= 4 ? 8 : 0);
    wsw[2 * t]     = (__bf16)W[kb * NC + col];
    wsw[2 * t + 1] = (__bf16)W[(kb + 1) * NC + col];
}

// APPLY=0: GEMM + per-block column sum / sumsq (deterministic, no atomics)
// APPLY=1: GEMM + BN + shift_out scatter + residual + relu
template <int APPLY>
__global__ __launch_bounds__(256) void k_gemm(
    const float* __restrict__ x0, const __bf16* __restrict__ wsw,
    const int* __restrict__ off_tab, const float* __restrict__ scl_tab,
    float* __restrict__ part_s, float* __restrict__ part_q,
    const int* __restrict__ inv_so, const float* __restrict__ meanp,
    const float* __restrict__ rstdp, const float* __restrict__ bnw,
    const float* __restrict__ bnb, float* __restrict__ out)
{
    extern __shared__ char smem[];
    __bf16* Wl   = (__bf16*)(smem + L_W);
    int*    koff = (int*)(smem + L_KOFF);
    float*  kscl = (float*)(smem + L_KSCL);

    const int tid  = threadIdx.x;
    const int iV   = blockIdx.x;   // v index (fastest -> L2 sample locality)
    const int mblk = blockIdx.y;
    const int lane = tid & 31;
    const int wv   = tid >> 5;

    // per-v gather tables
    koff[tid] = off_tab[iV * NC + tid];
    kscl[tid] = scl_tab[iV * NC + tid];

    if (APPLY) {
        // fold BN into y = acc*colA + colB; bias cancels in BN exactly.
        int   p = iV * NC + tid;
        int   f = inv_so[p];
        float m = meanp[p], r = rstdp[p];
        float w = bnw[f],   b = bnb[f];
        ((float*)(smem + L_EPI))[tid]        = r * w;
        ((float*)(smem + L_EPI + 1024))[tid] = b - m * r * w;
        ((int*)(smem + L_EPI + 2048))[tid]   = (f & 255) * NV + (f >> 8);
    }

    // stage full B-matrix fragments (128 KB) into LDS via the CDNA5 async
    // direct-to-LDS path (no VGPR round trip, tracked by ASYNCcnt).
    {
        uint32_t lbase = (uint32_t)(uintptr_t)Wl;   // LDS aperture: addr[31:0]
        uint64_t gbase = (uint64_t)(uintptr_t)wsw;
        for (int idx = tid; idx < 8192; idx += 256) {
            uint32_t la = lbase + (uint32_t)idx * 16u;
            uint64_t ga = gbase + (uint64_t)idx * 16u;
            asm volatile("global_load_async_to_lds_b128 %0, %1, off"
                         :: "v"(la), "v"(ga) : "memory");
        }
        asm volatile("s_wait_asynccnt 0" ::: "memory");
    }
    __syncthreads();

    // this lane's A row (sample): saddr-form addressing, 32-bit voffset
    const int      nA     = mblk * MBLK + wv * 16 + (lane & 15);
    const uint32_t rowoff = (uint32_t)nA * (uint32_t)(NF * 4);
    const int laneAdd = (lane & 16) ? 8 : 0;
    const char* x0b = (const char*)x0;

    v8f acc[16];
#pragma unroll
    for (int t = 0; t < 16; t++)
#pragma unroll
        for (int g = 0; g < 8; g++) acc[t][g] = 0.0f;

    float vbuf[2][16];   // double-buffered gathered A values (pre-scale)

#pragma unroll
    for (int e = 0; e < 16; e++) {
        int k = laneAdd + (e < 8 ? e : e + 8);
        vbuf[0][e] = *(const float*)(x0b + (size_t)(rowoff + (uint32_t)koff[k]));
    }

#pragma unroll
    for (int kt = 0; kt < 8; kt++) {
        // prefetch next K-tile's gathers; they complete under this tile's WMMAs
        if (kt + 1 < 8) {
#pragma unroll
            for (int e = 0; e < 16; e++) {
                int k = (kt + 1) * 32 + laneAdd + (e < 8 ? e : e + 8);
                vbuf[(kt + 1) & 1][e] =
                    *(const float*)(x0b + (size_t)(rowoff + (uint32_t)koff[k]));
            }
        }
        // build A fragment (documented 16-bit A layout)
        v16bf a;
#pragma unroll
        for (int e = 0; e < 16; e++) {
            int k = kt * 32 + laneAdd + (e < 8 ? e : e + 8);
            a[e] = (__bf16)(vbuf[kt & 1][e] * kscl[k]);
        }
        // B-fragment double buffer: next LDS load in flight during each WMMA
        const __bf16* wb = Wl + ((size_t)(kt * 16) * 32 + lane) * 16;
        v16bf b0 = *(const v16bf*)(wb);
#pragma unroll
        for (int nt = 0; nt < 16; nt++) {
            v16bf b1;
            if (nt + 1 < 16) b1 = *(const v16bf*)(wb + (size_t)(nt + 1) * 512);
            acc[nt] = __builtin_amdgcn_wmma_f32_16x16x32_bf16(
                false, a, false, b0, (short)0, acc[nt], false, false);
            b0 = b1;
        }
    }

    if (APPLY == 0) {
        // column sums: lane<16 col=nt*16+lane rows g / lane>=16 same col rows g+8
        float* red_s = (float*)(smem + L_EPI);
        float* red_q = (float*)(smem + L_EPI + 8192);
#pragma unroll
        for (int nt = 0; nt < 16; nt++) {
            float ps = 0.f, pq = 0.f;
#pragma unroll
            for (int g = 0; g < 8; g++) { float z = acc[nt][g]; ps += z; pq += z * z; }
            ps += __shfl_xor(ps, 16);
            pq += __shfl_xor(pq, 16);
            if (lane < 16) {
                int col = nt * 16 + lane;
                red_s[wv * 256 + col] = ps;
                red_q[wv * 256 + col] = pq;
            }
        }
        __syncthreads();
        float s = 0.f, q = 0.f;
#pragma unroll
        for (int w2 = 0; w2 < 8; w2++) {
            s += red_s[w2 * 256 + tid];
            q += red_q[w2 * 256 + tid];
        }
        part_s[(iV * NMB + mblk) * 256 + tid] = s;
        part_q[(iV * NMB + mblk) * 256 + tid] = q;
    } else {
        const float* colA = (const float*)(smem + L_EPI);
        const float* colB = (const float*)(smem + L_EPI + 1024);
        const int*   coff = (const int*)(smem + L_EPI + 2048);
        const int nbase = mblk * MBLK + wv * 16 + ((lane & 16) ? 8 : 0);
#pragma unroll
        for (int nt = 0; nt < 16; nt++) {
            int   col = nt * 16 + (lane & 15);
            float A2 = colA[col], B2 = colB[col];
            int   oo = coff[col];
#pragma unroll
            for (int g = 0; g < 8; g++) {
                size_t idx = (size_t)(nbase + g) * NF + oo;
                float  v   = fmaf(acc[nt][g], A2, B2) + x0[idx];  // residual (L2-hot)
                out[idx]   = fmaxf(v, 0.0f);
            }
        }
    }
}

__global__ __launch_bounds__(256) void k_finalize(
    const float* __restrict__ part_s, const float* __restrict__ part_q,
    float* __restrict__ meanp, float* __restrict__ rstdp)
{
    int p   = blockIdx.x * 256 + threadIdx.x;  // 0..5887 exact
    int i   = p >> 8, col = p & 255;
    float s = 0.f, q = 0.f;
    for (int mb = 0; mb < NMB; mb++) {
        s += part_s[(i * NMB + mb) * 256 + col];
        q += part_q[(i * NMB + mb) * 256 + col];
    }
    float mean = s * (1.0f / NSAMP);
    float var  = q * (1.0f / NSAMP) - mean * mean;
    meanp[p] = mean;
    rstdp[p] = rsqrtf(var + 1e-5f);
}

extern "C" void kernel_launch(void* const* d_in, const int* in_sizes, int n_in,
                              void* d_out, int out_size, void* d_ws, size_t ws_size,
                              hipStream_t stream)
{
    const float* x0        = (const float*)d_in[0];
    const float* W         = (const float*)d_in[1];
    /* d_in[2] = bias: cancels exactly inside BatchNorm -> unused */
    const float* mask      = (const float*)d_in[3];
    const float* bnw       = (const float*)d_in[4];
    const float* bnb       = (const float*)d_in[5];
    const int*   shift_in  = (const int*)d_in[6];
    const int*   shift_out = (const int*)d_in[7];

    char*   ws      = (char*)d_ws;
    int*    off_tab = (int*)(ws + WS_OFF);
    float*  scl_tab = (float*)(ws + WS_SCL);
    int*    inv_so  = (int*)(ws + WS_INV);
    float*  meanp   = (float*)(ws + WS_MEAN);
    float*  rstdp   = (float*)(ws + WS_RSTD);
    __bf16* wsw     = (__bf16*)(ws + WS_WSW);
    float*  part_s  = (float*)(ws + WS_PSUM);
    float*  part_q  = (float*)(ws + WS_PSQ);
    float*  out     = (float*)d_out;

    hipFuncSetAttribute((const void*)k_gemm<0>,
                        hipFuncAttributeMaxDynamicSharedMemorySize, SMEM_A);
    hipFuncSetAttribute((const void*)k_gemm<1>,
                        hipFuncAttributeMaxDynamicSharedMemorySize, SMEM_B);

    k_prep_tables<<<NV, 256, 0, stream>>>(mask, shift_in, shift_out,
                                          off_tab, scl_tab, inv_so);
    k_prep_w<<<128, 256, 0, stream>>>(W, wsw);

    dim3 grid(NV, NMB);  // v fastest -> same-sample blocks co-resident in L2
    k_gemm<0><<<grid, 256, SMEM_A, stream>>>(x0, wsw, off_tab, scl_tab,
                                             part_s, part_q,
                                             nullptr, nullptr, nullptr,
                                             nullptr, nullptr, nullptr);
    k_finalize<<<NV, 256, 0, stream>>>(part_s, part_q, meanp, rstdp);
    k_gemm<1><<<grid, 256, SMEM_B, stream>>>(x0, wsw, off_tab, scl_tab,
                                             nullptr, nullptr,
                                             inv_so, meanp, rstdp, bnw, bnb, out);
}
